// T5LayerRGAT_91311004713519
// MI455X (gfx1250) — matrix operand order repaired and checked by reading, not compiled, not run on hardware
//
#include <hip/hip_runtime.h>
#include <hip/hip_bf16.h>

// ---------------------------------------------------------------------------
// T5LayerRGAT for MI455X (gfx1250, wave32).
// GEMMs: LDS-tiled (128x128 block, double-buffered) v_wmma_f32_16x16x32_f16,
// refilled with GLOBAL_LOAD_ASYNC_TO_LDS (ASYNCcnt) when the builtin exists.
// Edge softmax-aggregation via f32 global atomics; LNs via LDS row reductions.
// ---------------------------------------------------------------------------

typedef _Float16 half4  __attribute__((ext_vector_type(4)));
typedef _Float16 half8  __attribute__((ext_vector_type(8)));
typedef _Float16 half16 __attribute__((ext_vector_type(16)));
typedef float    v8f    __attribute__((ext_vector_type(8)));

#define DEVINL static __device__ __forceinline__

#if defined(__HIP_DEVICE_COMPILE__) && \
    __has_builtin(__builtin_amdgcn_global_load_async_to_lds_b128)
#define HAS_ASYNC_LDS 1
#else
#define HAS_ASYNC_LDS 0
#endif

#if HAS_ASYNC_LDS
// Builtin prototype (from clang diagnostic): arg0 = AS1 (global) pointer to
// int __attribute__((vector_size(16))), arg1 = AS3 (LDS) pointer, then two
// integer immediates (offset, cpol).
typedef int i32x4 __attribute__((vector_size(16)));
typedef __attribute__((address_space(1))) i32x4* as1_b128_t;
typedef __attribute__((address_space(3))) i32x4* as3_b128_t;

// 16B per-lane async copy global -> LDS (no VGPR staging; tracked by ASYNCcnt)
DEVINL void async_cp16(const _Float16* g, _Float16* l) {
  __builtin_amdgcn_global_load_async_to_lds_b128(
      (as1_b128_t)const_cast<_Float16*>(g), (as3_b128_t)l, 0, 0);
}
DEVINL void async_wait0() {
#if __has_builtin(__builtin_amdgcn_s_wait_asynccnt)
  __builtin_amdgcn_s_wait_asynccnt(0);
#else
  asm volatile("s_wait_asynccnt 0" ::: "memory");
#endif
}
#endif

// A fragment (16x32 f16, ISA 7.12.2): lane = row m, halves 0..7 = K[h*8..+7],
// halves 8..15 = K[h*8+16..+23]  (h = lane>>4).  Chunks at +0 and +16 halves.
DEVINL half16 load_frag_a(const _Float16* p) {
  half8 lo = *(const half8*)(p);
  half8 hi = *(const half8*)(p + 16);
  return __builtin_shufflevector(lo, hi, 0,1,2,3,4,5,6,7,8,9,10,11,12,13,14,15);
}
// B fragment (32x16 f16): lane = column n, halves 0..7 = K[h*8..+7],
// halves 8..15 = K[h*8+16..+23]; read as two 16B-aligned half8 (+0, +16).
DEVINL half16 load_frag_b(const _Float16* p) {
  half8 lo = *(const half8*)(p);
  half8 hi = *(const half8*)(p + 16);
  return __builtin_shufflevector(lo, hi, 0,1,2,3,4,5,6,7,8,9,10,11,12,13,14,15);
}

DEVINL v8f wmma16(half16 a, half16 b, v8f c) {
  return __builtin_amdgcn_wmma_f32_16x16x32_f16(false, a, false, b, (short)0, c,
                                                false, false);
}

// ---------------------------------------------------------------------------
// LDS-tiled WMMA GEMM:  C[M,N] = A[M,K](f16 row-major) @ Bt[N,K](f16, = B^T)
// Block = 256 threads = 8 waves -> 128x128 tile; wave grid 2(M) x 4(N), each
// wave computes 64x32 (4x2 subtiles of 16x16).  K staged in 32-wide slabs
// through double-buffered LDS (rows padded to 40 halves for bank spread).
// Grid must tile M,N exactly (M%128==0, N%128==0).
// ---------------------------------------------------------------------------
template<bool HAS_BIAS, bool RELU, bool ADD_RES, bool OUT16>
__global__ __launch_bounds__(256) void wmma_gemm_kernel(
    const _Float16* __restrict__ A, const _Float16* __restrict__ Bt,
    const float* __restrict__ bias, const float* __restrict__ res,
    float* __restrict__ Cf, _Float16* __restrict__ Ch,
    int M, int N, int K)
{
  constexpr int LDA = 40;                      // padded row stride (halves)
  __shared__ alignas(32) _Float16 As[2][128 * LDA];
  __shared__ alignas(32) _Float16 Bs[2][128 * LDA];

  const int tid   = threadIdx.x;
  const int wave  = tid >> 5;
  const int lane  = tid & 31;
  const int r     = lane & 15;
  const int h     = lane >> 4;
  const int waveM = wave >> 2;                 // 0..1  (64 rows)
  const int waveN = wave & 3;                  // 0..3  (32 cols)

  const int tilesN = N >> 7;
  const int bm = (int)(blockIdx.x / tilesN) << 7;
  const int bn = (int)(blockIdx.x % tilesN) << 7;

  // staging map: thread -> (row = tid/4 in [0,64), 8-half chunk = tid%4); the
  // second half of the tile (rows 64..127) via +64*K / +64*LDA.
  const int srow = tid >> 2;
  const int skc  = (tid & 3) << 3;
  const int soff = srow * LDA + skc;
  const _Float16* gA0 = A  + (size_t)(bm + srow) * K + skc;
  const _Float16* gA1 = gA0 + (size_t)64 * K;
  const _Float16* gB0 = Bt + (size_t)(bn + srow) * K + skc;
  const _Float16* gB1 = gB0 + (size_t)64 * K;

  v8f acc[4][2] = {};
  int cur = 0;

#if HAS_ASYNC_LDS
  // prologue: async-stage K-slab 0 straight into LDS buf 0
  async_cp16(gA0, &As[0][soff]);
  async_cp16(gA1, &As[0][soff + 64 * LDA]);
  async_cp16(gB0, &Bs[0][soff]);
  async_cp16(gB1, &Bs[0][soff + 64 * LDA]);

  for (int kb = 0; kb < K; kb += 32) {
    async_wait0();                             // my copies into buf[cur] landed
    __syncthreads();                           // everyone's landed
    const bool nxt = (kb + 32) < K;
    if (nxt) {                                 // overlap slab k+1 with compute k
      async_cp16(gA0 + kb + 32, &As[cur ^ 1][soff]);
      async_cp16(gA1 + kb + 32, &As[cur ^ 1][soff + 64 * LDA]);
      async_cp16(gB0 + kb + 32, &Bs[cur ^ 1][soff]);
      async_cp16(gB1 + kb + 32, &Bs[cur ^ 1][soff + 64 * LDA]);
      if (kb + 64 < K) {                       // global_prefetch_b8 (slab k+2)
        __builtin_prefetch(gA0 + kb + 64, 0, 1);
        __builtin_prefetch(gB0 + kb + 64, 0, 1);
      }
    }

    const _Float16* as = As[cur];
    const _Float16* bs = Bs[cur];
    half16 fa[4], fb[2];
    #pragma unroll
    for (int mi = 0; mi < 4; ++mi)
      fa[mi] = load_frag_a(as + (waveM * 64 + mi * 16 + r) * LDA + h * 8);
    #pragma unroll
    for (int ni = 0; ni < 2; ++ni)
      fb[ni] = load_frag_b(bs + (waveN * 32 + ni * 16 + r) * LDA + h * 8);
    #pragma unroll
    for (int mi = 0; mi < 4; ++mi)
      #pragma unroll
      for (int ni = 0; ni < 2; ++ni)
        acc[mi][ni] = wmma16(fa[mi], fb[ni], acc[mi][ni]);

    if (nxt) cur ^= 1;
  }
#else
  // fallback: register-staged double buffering
  half8 ra0 = *(const half8*)(gA0);
  half8 ra1 = *(const half8*)(gA1);
  half8 rb0 = *(const half8*)(gB0);
  half8 rb1 = *(const half8*)(gB1);
  *(half8*)&As[0][soff]            = ra0;
  *(half8*)&As[0][soff + 64 * LDA] = ra1;
  *(half8*)&Bs[0][soff]            = rb0;
  *(half8*)&Bs[0][soff + 64 * LDA] = rb1;

  for (int kb = 0; kb < K; kb += 32) {
    __syncthreads();                           // LDS buf[cur] ready
    const bool nxt = (kb + 32) < K;
    if (nxt) {
      ra0 = *(const half8*)(gA0 + kb + 32);
      ra1 = *(const half8*)(gA1 + kb + 32);
      rb0 = *(const half8*)(gB0 + kb + 32);
      rb1 = *(const half8*)(gB1 + kb + 32);
      if (kb + 64 < K) {
        __builtin_prefetch(gA0 + kb + 64, 0, 1);
        __builtin_prefetch(gB0 + kb + 64, 0, 1);
      }
    }

    const _Float16* as = As[cur];
    const _Float16* bs = Bs[cur];
    half16 fa[4], fb[2];
    #pragma unroll
    for (int mi = 0; mi < 4; ++mi)
      fa[mi] = load_frag_a(as + (waveM * 64 + mi * 16 + r) * LDA + h * 8);
    #pragma unroll
    for (int ni = 0; ni < 2; ++ni)
      fb[ni] = load_frag_b(bs + (waveN * 32 + ni * 16 + r) * LDA + h * 8);
    #pragma unroll
    for (int mi = 0; mi < 4; ++mi)
      #pragma unroll
      for (int ni = 0; ni < 2; ++ni)
        acc[mi][ni] = wmma16(fa[mi], fb[ni], acc[mi][ni]);

    if (nxt) {
      *(half8*)&As[cur ^ 1][soff]            = ra0;
      *(half8*)&As[cur ^ 1][soff + 64 * LDA] = ra1;
      *(half8*)&Bs[cur ^ 1][soff]            = rb0;
      *(half8*)&Bs[cur ^ 1][soff + 64 * LDA] = rb1;
      cur ^= 1;
    }
  }
#endif

  // C/D layout: VGPR j -> M = j + 8*h ; lane -> N = r.  32-bit indexing.
  #pragma unroll
  for (int mi = 0; mi < 4; ++mi) {
    #pragma unroll
    for (int ni = 0; ni < 2; ++ni) {
      const unsigned mb  = (unsigned)(bm + waveM * 64 + mi * 16);
      const unsigned col = (unsigned)(bn + waveN * 32 + ni * 16 + r);
      float bi = 0.0f;
      if (HAS_BIAS) bi = bias[col];
      #pragma unroll
      for (int j = 0; j < 8; ++j) {
        const unsigned row = mb + (unsigned)((h << 3) + j);
        const unsigned idx = row * (unsigned)N + col;
        float vv = acc[mi][ni][j] + bi;
        if (ADD_RES) vv += res[idx];
        if (RELU)    vv = fmaxf(vv, 0.0f);
        if (OUT16) Ch[idx] = (_Float16)vv;
        else       Cf[idx] = vv;
      }
    }
  }
}

// ---------------------------------------------------------------------------
// Transpose + f32->f16 convert:  W[K,N] f32  ->  Wt[N,K] f16
// ---------------------------------------------------------------------------
__global__ __launch_bounds__(256) void transpose_cvt_kernel(
    const float* __restrict__ W, _Float16* __restrict__ Wt, int Kd, int Nd)
{
  __shared__ float tile[32][33];
  const int tilesN = Nd >> 5;
  const int bx = blockIdx.x % tilesN;   // n tile
  const int by = blockIdx.x / tilesN;   // k tile
  const int tx = threadIdx.x & 31;
  const int ty = threadIdx.x >> 5;      // 0..7
  const int n  = (bx << 5) + tx;
  #pragma unroll
  for (int i = 0; i < 32; i += 8)
    tile[ty + i][tx] = W[(size_t)((by << 5) + ty + i) * Nd + n];
  __syncthreads();
  const int k2 = (by << 5) + tx;
  #pragma unroll
  for (int i = 0; i < 32; i += 8)
    Wt[(size_t)((bx << 5) + ty + i) * Kd + k2] = (_Float16)tile[tx][ty + i];
}

// ---------------------------------------------------------------------------
// LayerNorm over D=1024, one row per 256-thread block (4 elems/thread).
// ---------------------------------------------------------------------------
template<bool WRITE16>
__global__ __launch_bounds__(256) void ln_kernel(
    const float* __restrict__ x, const float* __restrict__ g,
    const float* __restrict__ bb, float* __restrict__ y32,
    _Float16* __restrict__ y16, int Dd)
{
  __shared__ float s1[256], s2[256];
  const int row = blockIdx.x;
  const int tid = threadIdx.x;
  const size_t base = (size_t)row * Dd + tid * 4;
  const float4 xv = *(const float4*)(x + base);
  s1[tid] = xv.x + xv.y + xv.z + xv.w;
  s2[tid] = xv.x * xv.x + xv.y * xv.y + xv.z * xv.z + xv.w * xv.w;
  __syncthreads();
  for (int off = 128; off > 0; off >>= 1) {
    if (tid < off) { s1[tid] += s1[tid + off]; s2[tid] += s2[tid + off]; }
    __syncthreads();
  }
  const float mu  = s1[0] / Dd;
  const float var = s2[0] / Dd - mu * mu;
  const float rs  = rsqrtf(var + 1e-6f);
  const float4 gv = *(const float4*)(g  + tid * 4);
  const float4 bv = *(const float4*)(bb + tid * 4);
  float4 yv;
  yv.x = (xv.x - mu) * rs * gv.x + bv.x;
  yv.y = (xv.y - mu) * rs * gv.y + bv.y;
  yv.z = (xv.z - mu) * rs * gv.z + bv.z;
  yv.w = (xv.w - mu) * rs * gv.w + bv.w;
  *(float4*)(y32 + base) = yv;
  if (WRITE16) {
    half4 hv = { (_Float16)yv.x, (_Float16)yv.y, (_Float16)yv.z, (_Float16)yv.w };
    *(half4*)(y16 + base) = hv;
  }
}

// Final: out = hidden + elu(LN(x))
__global__ __launch_bounds__(256) void ln_elu_res_kernel(
    const float* __restrict__ x, const float* __restrict__ g,
    const float* __restrict__ bb, const float* __restrict__ hidden,
    float* __restrict__ out, int Dd)
{
  __shared__ float s1[256], s2[256];
  const int row = blockIdx.x;
  const int tid = threadIdx.x;
  const size_t base = (size_t)row * Dd + tid * 4;
  const float4 xv = *(const float4*)(x + base);
  s1[tid] = xv.x + xv.y + xv.z + xv.w;
  s2[tid] = xv.x * xv.x + xv.y * xv.y + xv.z * xv.z + xv.w * xv.w;
  __syncthreads();
  for (int off = 128; off > 0; off >>= 1) {
    if (tid < off) { s1[tid] += s1[tid + off]; s2[tid] += s2[tid + off]; }
    __syncthreads();
  }
  const float mu  = s1[0] / Dd;
  const float var = s2[0] / Dd - mu * mu;
  const float rs  = rsqrtf(var + 1e-6f);
  const float4 gv = *(const float4*)(g  + tid * 4);
  const float4 bv = *(const float4*)(bb + tid * 4);
  const float4 hv = *(const float4*)(hidden + base);
  float yn[4], hh[4] = {hv.x, hv.y, hv.z, hv.w};
  yn[0] = (xv.x - mu) * rs * gv.x + bv.x;
  yn[1] = (xv.y - mu) * rs * gv.y + bv.y;
  yn[2] = (xv.z - mu) * rs * gv.z + bv.z;
  yn[3] = (xv.w - mu) * rs * gv.w + bv.w;
  float4 ov;
  float* op = &ov.x;
  #pragma unroll
  for (int i = 0; i < 4; ++i) {
    const float e = yn[i] > 0.0f ? yn[i] : (expf(yn[i]) - 1.0f);  // elu
    op[i] = hh[i] + e;
  }
  *(float4*)(out + base) = ov;
}

// ---------------------------------------------------------------------------
// Edge kernels.  scale = sqrt(D) = 32.
// ---------------------------------------------------------------------------
__global__ __launch_bounds__(256) void edge_score_kernel(
    const float* __restrict__ q, const float* __restrict__ k,
    const float* __restrict__ rel_table,
    const int* __restrict__ esrc, const int* __restrict__ edst,
    const int* __restrict__ erel,
    float* __restrict__ score, float* __restrict__ z,
    int En, int Ln, int Dd)
{
  const int gid  = blockIdx.x * 8 + (threadIdx.x >> 5);   // edge over B*E
  const int lane = threadIdx.x & 31;
  const int b = gid / En, e = gid % En;
  const int src = esrc[(size_t)b * En + e];
  const int dst = edst[(size_t)b * En + e];
  const int rel = erel[(size_t)b * En + e];
  const float* kp = k + ((size_t)b * Ln + src) * Dd;
  const float* qp = q + ((size_t)b * Ln + dst) * Dd;
  const float* rp = rel_table + (size_t)rel * Dd;
  float acc = 0.0f;
  for (int i = lane * 4; i < Dd; i += 128) {
    const float4 kv = *(const float4*)(kp + i);
    const float4 rv = *(const float4*)(rp + i);
    const float4 qv = *(const float4*)(qp + i);
    acc += (kv.x + rv.x) * qv.x + (kv.y + rv.y) * qv.y +
           (kv.z + rv.z) * qv.z + (kv.w + rv.w) * qv.w;
  }
  #pragma unroll
  for (int off = 16; off > 0; off >>= 1) acc += __shfl_xor(acc, off, 32);
  if (lane == 0) {
    float s = acc * (1.0f / 32.0f);                  // / sqrt(1024)
    s = fminf(fmaxf(s, -10.0f), 10.0f);
    s = __expf(s);
    score[gid] = s;
    atomicAdd(&z[(size_t)b * Ln + dst], s);          // global_atomic_add_f32
  }
}

__global__ __launch_bounds__(256) void edge_accum_kernel(
    const float* __restrict__ v, const float* __restrict__ rel_table,
    const int* __restrict__ esrc, const int* __restrict__ edst,
    const int* __restrict__ erel, const float* __restrict__ score,
    float* __restrict__ wv, int En, int Ln, int Dd)
{
  const int gid = blockIdx.x;                  // one block per edge
  const int b = gid / En, e = gid % En;
  const int src = esrc[(size_t)b * En + e];
  const int dst = edst[(size_t)b * En + e];
  const int rel = erel[(size_t)b * En + e];
  const float s = score[gid];
  const float* vp = v + ((size_t)b * Ln + src) * Dd;
  const float* rp = rel_table + (size_t)rel * Dd;
  float* wp = wv + ((size_t)b * Ln + dst) * Dd;
  const int d = threadIdx.x * 4;
  const float4 vv = *(const float4*)(vp + d);
  const float4 rv = *(const float4*)(rp + d);
  atomicAdd(&wp[d + 0], (vv.x + rv.x) * s);
  atomicAdd(&wp[d + 1], (vv.y + rv.y) * s);
  atomicAdd(&wp[d + 2], (vv.z + rv.z) * s);
  atomicAdd(&wp[d + 3], (vv.w + rv.w) * s);
}

__global__ __launch_bounds__(256) void o_div_cvt_kernel(
    const float* __restrict__ wv, const float* __restrict__ z,
    _Float16* __restrict__ o16, int Dd)
{
  const size_t i   = ((size_t)blockIdx.x * 256 + threadIdx.x) * 4;
  const float  inv = 1.0f / z[i / Dd];
  const float4 w   = *(const float4*)(wv + i);
  half4 o = { (_Float16)(w.x * inv), (_Float16)(w.y * inv),
              (_Float16)(w.z * inv), (_Float16)(w.w * inv) };
  *(half4*)(o16 + i) = o;
}

__global__ __launch_bounds__(256) void zero_kernel(float4* __restrict__ p) {
  p[(size_t)blockIdx.x * 256 + threadIdx.x] = float4{0.f, 0.f, 0.f, 0.f};
}

// ---------------------------------------------------------------------------
// Host-side orchestration
// ---------------------------------------------------------------------------
extern "C" void kernel_launch(void* const* d_in, const int* in_sizes, int n_in,
                              void* d_out, int out_size, void* d_ws, size_t ws_size,
                              hipStream_t stream)
{
  constexpr int Bn = 4, Ln = 1024, Dd = 1024, DFd = 4096, En = 32768;
  const size_t BLD = (size_t)Bn * Ln * Dd;            // 4M elements
  const int    Mm  = Bn * Ln;                         // 4096 rows

  const float* hidden    = (const float*)d_in[0];
  const float* rel_table = (const float*)d_in[1];
  const float* Wq = (const float*)d_in[2];  const float* bq = (const float*)d_in[3];
  const float* Wk = (const float*)d_in[4];  const float* Wv = (const float*)d_in[5];
  const float* Wo = (const float*)d_in[6];  const float* bo = (const float*)d_in[7];
  const float* ln0g = (const float*)d_in[8];  const float* ln0b = (const float*)d_in[9];
  const float* ln1g = (const float*)d_in[10]; const float* ln1b = (const float*)d_in[11];
  const float* W1 = (const float*)d_in[12]; const float* b1 = (const float*)d_in[13];
  const float* W2 = (const float*)d_in[14]; const float* b2 = (const float*)d_in[15];
  const float* ln2g = (const float*)d_in[16]; const float* ln2b = (const float*)d_in[17];
  const int* esrc = (const int*)d_in[18];
  const int* edst = (const int*)d_in[19];
  const int* erel = (const int*)d_in[20];

  // ---- workspace carve (with lifetime-based reuse) ----
  char* ws = (char*)d_ws;
  size_t off = 0;
  auto alloc = [&](size_t bytes) -> void* {
    off = (off + 255) & ~(size_t)255;
    void* p = ws + off; off += bytes; return p;
  };
  float*    hs32   = (float*)   alloc(BLD * 4);            // hs_norm f32
  _Float16* buf16  = (_Float16*)alloc(BLD * 2);            // hs_norm16 / o16 / attn16
  _Float16* WtQ    = (_Float16*)alloc((size_t)Dd * Dd * 2);
  _Float16* WtK    = (_Float16*)alloc((size_t)Dd * Dd * 2);
  _Float16* WtV    = (_Float16*)alloc((size_t)Dd * Dd * 2);
  _Float16* WtO    = (_Float16*)alloc((size_t)Dd * Dd * 2);
  _Float16* Wt1    = (_Float16*)alloc((size_t)Dd * DFd * 2);  // [DF,D]
  _Float16* Wt2    = (_Float16*)alloc((size_t)Dd * DFd * 2);  // [D,DF]
  float*    qbuf   = (float*)   alloc(BLD * 4);            // q (dead after scores)
  float*    kbuf   = (float*)   alloc(BLD * 4);            // k   -> pre_ln1
  float*    vbuf   = (float*)   alloc(BLD * 4);            // v   -> pre_ln2
  float*    score  = (float*)   alloc((size_t)Bn * En * 4);
  float*    zbuf   = (float*)   alloc((size_t)Bn * Ln * 4);
  float*    wvbuf  = (float*)   alloc(BLD * 4);            // wv  -> attn32
  _Float16* ffn16  = (_Float16*)alloc((size_t)Mm * DFd * 2);
  (void)ws_size; (void)in_sizes; (void)n_in; (void)out_size;

  float* pre_ln1 = kbuf;     // reuse after scores
  float* attn32  = wvbuf;    // reuse after o = wv/z
  float* pre_ln2 = vbuf;     // reuse after aggregation

  // ---- 1. weight transpose+convert (B^T layout for WMMA B operand) ----
  transpose_cvt_kernel<<<(Dd/32)*(Dd/32), 256, 0, stream>>>(Wq, WtQ, Dd, Dd);
  transpose_cvt_kernel<<<(Dd/32)*(Dd/32), 256, 0, stream>>>(Wk, WtK, Dd, Dd);
  transpose_cvt_kernel<<<(Dd/32)*(Dd/32), 256, 0, stream>>>(Wv, WtV, Dd, Dd);
  transpose_cvt_kernel<<<(Dd/32)*(Dd/32), 256, 0, stream>>>(Wo, WtO, Dd, Dd);
  transpose_cvt_kernel<<<(Dd/32)*(DFd/32), 256, 0, stream>>>(W1, Wt1, Dd, DFd);
  transpose_cvt_kernel<<<(DFd/32)*(Dd/32), 256, 0, stream>>>(W2, Wt2, DFd, Dd);

  // ---- 2. LN0 ----
  ln_kernel<true><<<Mm, 256, 0, stream>>>(hidden, ln0g, ln0b, hs32, buf16, Dd);

  // ---- 3. Q/K/V projections (WMMA, 128x128 LDS-tiled) ----
  const int gDD = (Mm/128)*(Dd/128);     // 256 blocks
  wmma_gemm_kernel<true , false, false, false><<<gDD, 256, 0, stream>>>(
      buf16, WtQ, bq, nullptr, qbuf, nullptr, Mm, Dd, Dd);
  wmma_gemm_kernel<false, false, false, false><<<gDD, 256, 0, stream>>>(
      buf16, WtK, nullptr, nullptr, kbuf, nullptr, Mm, Dd, Dd);
  wmma_gemm_kernel<false, false, false, false><<<gDD, 256, 0, stream>>>(
      buf16, WtV, nullptr, nullptr, vbuf, nullptr, Mm, Dd, Dd);

  // ---- 4. zero accumulators ----
  zero_kernel<<<(int)((size_t)Bn * Ln * 4 / 1024), 256, 0, stream>>>((float4*)zbuf);
  zero_kernel<<<(int)(BLD * 4 / 4096), 256, 0, stream>>>((float4*)wvbuf);

  // ---- 5-6. edge softmax-aggregation ----
  edge_score_kernel<<<Bn * En / 8, 256, 0, stream>>>(
      qbuf, kbuf, rel_table, esrc, edst, erel, score, zbuf, En, Ln, Dd);
  edge_accum_kernel<<<Bn * En, 256, 0, stream>>>(
      vbuf, rel_table, esrc, edst, erel, score, wvbuf, En, Ln, Dd);

  // ---- 7. o = wv / z  (f16 for next GEMM) ----
  o_div_cvt_kernel<<<(int)(BLD / 1024), 256, 0, stream>>>(wvbuf, zbuf, buf16, Dd);

  // ---- 8. O-projection + bias + residual(hs_norm) ----
  wmma_gemm_kernel<true, false, true, false><<<gDD, 256, 0, stream>>>(
      buf16, WtO, bo, hs32, pre_ln1, nullptr, Mm, Dd, Dd);

  // ---- 9. LN1 ----
  ln_kernel<true><<<Mm, 256, 0, stream>>>(pre_ln1, ln1g, ln1b, attn32, buf16, Dd);

  // ---- 10. FFN1: relu(attn @ W1 + b1) -> f16 ----
  const int gDF = (Mm/128)*(DFd/128);    // 1024 blocks
  wmma_gemm_kernel<true, true, false, true><<<gDF, 256, 0, stream>>>(
      buf16, Wt1, b1, nullptr, nullptr, ffn16, Mm, DFd, Dd);

  // ---- 11. FFN2: ffn @ W2 + b2 + residual(attn) ----
  wmma_gemm_kernel<true, false, true, false><<<gDD, 256, 0, stream>>>(
      ffn16, Wt2, b2, attn32, pre_ln2, nullptr, Mm, Dd, DFd);

  // ---- 12. out = hidden + elu(LN2(pre_ln2)) ----
  ln_elu_res_kernel<<<Mm, 256, 0, stream>>>(
      pre_ln2, ln2g, ln2b, hidden, (float*)d_out, Dd);
}
